// MultiHeadAttention_53858889891941
// MI455X (gfx1250) — compile-verified, matrix operand order checked by reading, not run on hardware
//
#include <hip/hip_runtime.h>
#include <hip/hip_bf16.h>
#include <math.h>

#define DEV __device__ __forceinline__

typedef __attribute__((ext_vector_type(16))) _Float16 v16h;
typedef __attribute__((ext_vector_type(8)))  float    v8f;
typedef __attribute__((ext_vector_type(4)))  unsigned u32x4;
typedef __attribute__((ext_vector_type(8)))  unsigned u32x8;

constexpr int NCTX  = 2048;
constexpr int NEMBD = 1024;
constexpr int NHEAD = 16;
constexpr int HDIM  = 64;
constexpr int MROWS = 4 * NCTX;          // B * L = 8192

// ---------------------------------------------------------------------------
// CDNA5 async Global->LDS copy (ASYNCcnt path), 16B per lane.
// LDS byte offset = low 32 bits of the flat shared-memory address.
// ---------------------------------------------------------------------------
DEV void async_copy_b128(void* lds_dst, const void* gsrc) {
  const unsigned lds = (unsigned)(uintptr_t)lds_dst;
  asm volatile("global_load_async_to_lds_b128 %0, %1, off"
               :: "v"(lds), "v"(gsrc) : "memory");
}
DEV void wait_async() { asm volatile("s_wait_asynccnt 0" ::: "memory"); }

// ---------------------------------------------------------------------------
// CDNA5 Tensor Data Mover: 2-D tile Global->LDS DMA (TENSORcnt path).
// D# per ISA 08 (§8.3 group0, §8.4 group1); VADDR2/3 omitted (2-D tensor).
// pad_interval_code: pad every 2^(code+1) DWORDs; pad_amount_code: (code+1)
// DWORDs of LDS padding -- reproduces the bank-conflict row pad in hardware.
// ---------------------------------------------------------------------------
DEV void tdm_load_2d(unsigned lds_addr, const void* gaddr,
                     unsigned tensor_w, unsigned tensor_h, unsigned stride_elems,
                     unsigned tile_w, unsigned tile_h,
                     unsigned pad_interval_code, unsigned pad_amount_code) {
  const unsigned long long ga = (unsigned long long)(uintptr_t)gaddr;
  u32x4 g0;
  g0[0] = 1u;                                              // count=1 (valid), user mode
  g0[1] = lds_addr;                                        // LDS byte address
  g0[2] = (unsigned)(ga & 0xFFFFFFFFu);                    // global_addr[31:0]
  g0[3] = (unsigned)((ga >> 32) & 0x01FFFFFFu)             // global_addr[56:32]
          | 0x80000000u;                                   // type=2 ("image")
  u32x8 g1;
  g1[0] = (1u << 16)                                       // data_size=1 -> 2 bytes
        | (1u << 20)                                       // pad_enable
        | (pad_interval_code << 22)
        | (pad_amount_code << 25);
  g1[1] = (tensor_w & 0xFFFFu) << 16;                      // tensor_dim0[15:0]
  g1[2] = ((tensor_w >> 16) & 0xFFFFu)                     // tensor_dim0[31:16]
        | ((tensor_h & 0xFFFFu) << 16);                    // tensor_dim1[15:0]
  g1[3] = ((tensor_h >> 16) & 0xFFFFu)                     // tensor_dim1[31:16]
        | ((tile_w & 0xFFFFu) << 16);                      // tile_dim0
  g1[4] = (tile_h & 0xFFFFu);                              // tile_dim1 (tile_dim2=0)
  g1[5] = stride_elems;                                    // tensor_dim0_stride[31:0]
  g1[6] = 0;                                               // stride[47:32], dim1_stride
  g1[7] = 0;
  asm volatile("tensor_load_to_lds %0, %1" :: "s"(g0), "s"(g1) : "memory");
}
DEV void wait_tensor() { __builtin_amdgcn_s_wait_tensorcnt(0); }

// ---------------------------------------------------------------------------
// WMMA wrapper: D = A(16x32 f16) x B(32x16 f16) + C(16x16 f32)
// ---------------------------------------------------------------------------
DEV v8f wmma_f16(v16h a, v16h b, v8f c) {
  return __builtin_amdgcn_wmma_f32_16x16x32_f16(
      /*neg_a=*/false, a, /*neg_b=*/false, b,
      /*c_mod=*/(short)0, c, /*reuse_a=*/false, /*reuse_b=*/false);
}

// ---------------------------------------------------------------------------
// Fragment loaders per CDNA5 ISA 7.12.2 (wave32).
// A fragment 16x32 (MxK): lane&15 = M row; lane>>4 selects K half; VGPR j
// holds K pair: k = 8*(lane>>4) + (j<4 ? 0 : 16) + 2*(j&3).
// ---------------------------------------------------------------------------
DEV v16h load_a(const _Float16* p, int row0, int col0, int ld) {
  const int lane = (int)(threadIdx.x & 31u);
  const _Float16* r = p + (size_t)(row0 + (lane & 15)) * ld + col0 + 8 * (lane >> 4);
  v16h a;
#pragma unroll
  for (int j = 0; j < 8; ++j) {
    const int k = ((j < 4) ? 0 : 16) + 2 * (j & 3);
    a[2 * j]     = r[k];
    a[2 * j + 1] = r[k + 1];
  }
  return a;
}

// B fragment 32x16 (KxN): lane&15 = N column; lane>>4 selects K half.
DEV v16h load_b(const _Float16* p, int k0, int n0, int ld) {
  const int lane = (int)(threadIdx.x & 31u);
  const _Float16* c = p + (size_t)(k0 + 8 * (lane >> 4)) * ld + n0 + (lane & 15);
  v16h b;
#pragma unroll
  for (int j = 0; j < 8; ++j) {
    const int k = ((j < 4) ? 0 : 16) + 2 * (j & 3);
    b[2 * j]     = c[(size_t)k * ld];
    b[2 * j + 1] = c[(size_t)(k + 1) * ld];
  }
  return b;
}

// Row reductions across the 16 lanes sharing a C-fragment row (wave32).
DEV float rowmax16(float v) {
#pragma unroll
  for (int m = 8; m >= 1; m >>= 1) v = fmaxf(v, __shfl_xor(v, m, 32));
  return v;
}
DEV float rowsum16(float v) {
#pragma unroll
  for (int m = 8; m >= 1; m >>= 1) v += __shfl_xor(v, m, 32);
  return v;
}

// ---------------------------------------------------------------------------
// f32 -> f16 conversion (4 elems/thread, n multiple of 4)
// ---------------------------------------------------------------------------
__global__ void __launch_bounds__(256)
cvt_f32_f16_kernel(const float* __restrict__ src, _Float16* __restrict__ dst, int n) {
  const int i = (int)(blockIdx.x * 256 + threadIdx.x) * 4;
  if (i < n) {
    const float4 v = *(const float4*)(src + i);
    dst[i + 0] = (_Float16)v.x;
    dst[i + 1] = (_Float16)v.y;
    dst[i + 2] = (_Float16)v.z;
    dst[i + 3] = (_Float16)v.w;
  }
}

// ---------------------------------------------------------------------------
// Tiled WMMA GEMM: C[M,N] = A[M,K] * B[K,N] + bias[N]
// Block: 256 threads / 8 waves, tile 128x64, K step 32.
// Tile staging via TDM (tensor_load_to_lds), double-buffered: wave 0 issues
// the DMA for tile t+1, all waves compute WMMAs on tile t.
// ---------------------------------------------------------------------------
template <bool OUT_F32>
__global__ void __launch_bounds__(256)
gemm_f16_kernel(const _Float16* __restrict__ A, const _Float16* __restrict__ B,
                const float* __restrict__ bias, void* __restrict__ Cp,
                int M, int N, int K) {
  constexpr int BM = 128, BN = 64, BK = 32;
  __shared__ _Float16 As[2][BM][BK + 8];   // 20.0 KB; TDM pads rows by 4 DWORDs
  __shared__ _Float16 Bs[2][BK][BN + 8];   //  9.0 KB

  const int bm   = (int)blockIdx.x * BM;
  const int bn   = (int)blockIdx.y * BN;
  const int tid  = (int)threadIdx.x;
  const int wave = tid >> 5;
  const int lane = tid & 31;

  auto stage = [&](int k0, int buf) {
    if (wave == 0) {
      // A tile: 128 rows x 32 halves; row = 16 DWORDs -> pad code 3, +4 DWORDs
      tdm_load_2d((unsigned)(uintptr_t)&As[buf][0][0],
                  A + (size_t)bm * K + k0,
                  (unsigned)(K - k0), (unsigned)(M - bm), (unsigned)K,
                  BK, BM, /*interval 16dw*/3, /*amount 4dw*/3);
      // B tile: 32 rows x 64 halves; row = 32 DWORDs -> pad code 4, +4 DWORDs
      tdm_load_2d((unsigned)(uintptr_t)&Bs[buf][0][0],
                  B + (size_t)k0 * N + bn,
                  (unsigned)(N - bn), (unsigned)(K - k0), (unsigned)N,
                  BN, BK, /*interval 32dw*/4, /*amount 4dw*/3);
    }
  };

  v8f acc[4] = {};

  stage(0, 0);
  if (wave == 0) wait_tensor();
  __syncthreads();

  for (int k0 = 0; k0 < K; k0 += BK) {
    const int cur = (k0 / BK) & 1;
    if (k0 + BK < K) stage(k0 + BK, cur ^ 1);   // DMA overlaps compute

    const v16h a = load_a(&As[cur][0][0], wave * 16, 0, BK + 8);
#pragma unroll
    for (int nt = 0; nt < 4; ++nt) {
      const v16h b = load_b(&Bs[cur][0][0], 0, nt * 16, BN + 8);
      acc[nt] = wmma_f16(a, b, acc[nt]);
    }

    if (wave == 0) wait_tensor();   // next tile landed
    __syncthreads();                // visible to all waves
  }

  const int nl = lane & 15, mh = lane >> 4;
#pragma unroll
  for (int nt = 0; nt < 4; ++nt) {
    const int gn = bn + nt * 16 + nl;
    const float bv = bias[gn];
#pragma unroll
    for (int r = 0; r < 8; ++r) {
      const int gm = bm + wave * 16 + r + 8 * mh;
      const float val = acc[nt][r] + bv;
      if (OUT_F32) ((float*)Cp)[(size_t)gm * N + gn] = val;
      else         ((_Float16*)Cp)[(size_t)gm * N + gn] = (_Float16)val;
    }
  }
}

// ---------------------------------------------------------------------------
// RoPE + head split: qkv[B*L, 3D] f16 -> Q,K (rotated), V in [B,H,L,Dh] f16.
// ---------------------------------------------------------------------------
__global__ void __launch_bounds__(256)
rope_pack_kernel(const _Float16* __restrict__ qkv, _Float16* __restrict__ Q,
                 _Float16* __restrict__ K, _Float16* __restrict__ V) {
  const int idx  = (int)(blockIdx.x * 256 + threadIdx.x);
  const int pair = idx & 31;
  const int h    = (idx >> 5) & (NHEAD - 1);
  const int l    = (idx >> 9) & (NCTX - 1);
  const int b    = idx >> 20;

  const _Float16* src = qkv + ((size_t)b * NCTX + l) * (3 * NEMBD);
  const int base = h * HDIM + 2 * pair;

  const float inv = __powf(10000.0f, -(float)(2 * pair) * (1.0f / (float)HDIM));
  float s, c;
  __sincosf((float)l * inv, &s, &c);

  const float q1 = (float)src[base], q2 = (float)src[base + 1];
  const float k1 = (float)src[NEMBD + base], k2 = (float)src[NEMBD + base + 1];

  const size_t dst = ((size_t)(b * NHEAD + h) * NCTX + l) * HDIM + 2 * pair;
  Q[dst]     = (_Float16)(q1 * c - q2 * s);
  Q[dst + 1] = (_Float16)(q1 * s + q2 * c);
  K[dst]     = (_Float16)(k1 * c - k2 * s);
  K[dst + 1] = (_Float16)(k1 * s + k2 * c);
  V[dst]     = src[2 * NEMBD + base];
  V[dst + 1] = src[2 * NEMBD + base + 1];
}

// ---------------------------------------------------------------------------
// Flash attention (causal + key mask), online softmax, double-buffered
// async K/V staging (ASYNCcnt path).
// Grid: (B*H, L/128). Block: 256 threads / 8 waves.
// ---------------------------------------------------------------------------
__global__ void __launch_bounds__(256)
attn_kernel(const _Float16* __restrict__ Qg, const _Float16* __restrict__ Kg,
            const _Float16* __restrict__ Vg, const int* __restrict__ amask,
            _Float16* __restrict__ Og) {
  __shared__ _Float16 Ks[2][32][HDIM + 8];   // 9.0 KB
  __shared__ _Float16 Vs[2][32][HDIM + 8];   // 9.0 KB
  __shared__ _Float16 Ps[8][16][40];         // 10.0 KB, per-wave P staging

  const int bh = (int)blockIdx.x;            // b*H + h
  const int b  = bh >> 4;
  const int h  = bh & (NHEAD - 1);
  const int qb = (int)blockIdx.y * 128;
  const int tid = (int)threadIdx.x, wave = tid >> 5, lane = tid & 31;
  const int nl = lane & 15, mh = lane >> 4;

  const _Float16* q = Qg + (size_t)bh * NCTX * HDIM;
  const _Float16* k = Kg + (size_t)bh * NCTX * HDIM;
  const _Float16* v = Vg + (size_t)bh * NCTX * HDIM;

  const int kr = tid >> 3, kc = (tid & 7) * 8;   // staging slot (32x64 tile)
  auto stage = [&](int j0, int buf) {
    async_copy_b128(&Ks[buf][kr][kc], k + (size_t)(j0 + kr) * HDIM + kc);
    async_copy_b128(&Vs[buf][kr][kc], v + (size_t)(j0 + kr) * HDIM + kc);
  };

  const int qrow0 = qb + wave * 16;
  const v16h aq0 = load_a(q, qrow0, 0,  HDIM);   // q tile, K-dims 0..31
  const v16h aq1 = load_a(q, qrow0, 32, HDIM);   // q tile, K-dims 32..63

  v8f o[4] = {};
  float mrow[8], lrow[8];
#pragma unroll
  for (int r = 0; r < 8; ++r) { mrow[r] = -3.0e38f; lrow[r] = 0.0f; }

  const int jend = qb + 128;                     // causal upper bound for block
  stage(0, 0);
  wait_async();
  __syncthreads();

  for (int j0 = 0; j0 < jend; j0 += 32) {
    const int cur = (j0 >> 5) & 1;
    if (j0 + 32 < jend) stage(j0 + 32, cur ^ 1);

    if (j0 <= qrow0 + 15) {                      // block has relevant keys
      // S = Q (16x64) * K^T (64x32): two 16x16 tiles, 2 WMMAs each.
      v8f s0 = {}, s1 = {};
      s0 = wmma_f16(aq0, load_a(&Ks[cur][0][0], 0,  0,  HDIM + 8), s0);
      s0 = wmma_f16(aq1, load_a(&Ks[cur][0][0], 0,  32, HDIM + 8), s0);
      s1 = wmma_f16(aq0, load_a(&Ks[cur][0][0], 16, 0,  HDIM + 8), s1);
      s1 = wmma_f16(aq1, load_a(&Ks[cur][0][0], 16, 32, HDIM + 8), s1);

      const int kj0 = j0 + nl, kj1 = kj0 + 16;
      const bool keep0 = amask[b * NCTX + kj0] != 0;
      const bool keep1 = amask[b * NCTX + kj1] != 0;

#pragma unroll
      for (int r = 0; r < 8; ++r) {
        const int qi = qrow0 + r + 8 * mh;
        const float e0 = (keep0 && kj0 <= qi) ? s0[r] * 0.125f : -3.0e38f;
        const float e1 = (keep1 && kj1 <= qi) ? s1[r] * 0.125f : -3.0e38f;
        const float mnew  = fmaxf(mrow[r], rowmax16(fmaxf(e0, e1)));
        const float alpha = __expf(mrow[r] - mnew);
        const float p0 = __expf(e0 - mnew);
        const float p1 = __expf(e1 - mnew);
        lrow[r] = lrow[r] * alpha + rowsum16(p0 + p1);
        mrow[r] = mnew;
#pragma unroll
        for (int nt = 0; nt < 4; ++nt) o[nt][r] *= alpha;
        Ps[wave][r + 8 * mh][nl]      = (_Float16)p0;
        Ps[wave][r + 8 * mh][nl + 16] = (_Float16)p1;
      }
      asm volatile("s_wait_dscnt 0" ::: "memory");  // P visible to own wave

      // O += P (16x32) * V (32x64)
      const v16h ap = load_a(&Ps[wave][0][0], 0, 0, 40);
#pragma unroll
      for (int nt = 0; nt < 4; ++nt) {
        const v16h bv = load_b(&Vs[cur][0][0], 0, nt * 16, HDIM + 8);
        o[nt] = wmma_f16(ap, bv, o[nt]);
      }
    }

    wait_async();
    __syncthreads();
  }

  // normalize and write [B, L, D] f16
#pragma unroll
  for (int r = 0; r < 8; ++r) {
    const float invl = 1.0f / lrow[r];
    const int row = qrow0 + r + 8 * mh;
    _Float16* dst = Og + ((size_t)b * NCTX + row) * NEMBD + h * HDIM;
#pragma unroll
    for (int nt = 0; nt < 4; ++nt)
      dst[nt * 16 + nl] = (_Float16)(o[nt][r] * invl);
  }
}

// ---------------------------------------------------------------------------
extern "C" void kernel_launch(void* const* d_in, const int* in_sizes, int n_in,
                              void* d_out, int out_size, void* d_ws, size_t ws_size,
                              hipStream_t stream) {
  (void)in_sizes; (void)n_in; (void)out_size; (void)ws_size;

  const float* x     = (const float*)d_in[0];
  const float* Wqkv  = (const float*)d_in[1];
  const float* bqkv  = (const float*)d_in[2];
  const float* Wproj = (const float*)d_in[3];
  const float* bproj = (const float*)d_in[4];
  const int*   amask = (const int*)d_in[5];

  char* ws = (char*)d_ws;
  size_t off = 0;
  auto alloc = [&](size_t bytes) -> char* {
    char* p = ws + off;
    off = (off + bytes + 255) & ~(size_t)255;
    return p;
  };

  _Float16* xh    = (_Float16*)alloc((size_t)MROWS * NEMBD * 2);       // 16 MB
  _Float16* wqkvh = (_Float16*)alloc((size_t)NEMBD * 3 * NEMBD * 2);   //  6 MB
  _Float16* wprjh = (_Float16*)alloc((size_t)NEMBD * NEMBD * 2);       //  2 MB
  _Float16* qkvh  = (_Float16*)alloc((size_t)MROWS * 3 * NEMBD * 2);   // 48 MB
  _Float16* qh    = (_Float16*)alloc((size_t)MROWS * NEMBD * 2);       // 16 MB
  _Float16* kh    = (_Float16*)alloc((size_t)MROWS * NEMBD * 2);       // 16 MB
  _Float16* vh    = (_Float16*)alloc((size_t)MROWS * NEMBD * 2);       // 16 MB
  _Float16* oh    = (_Float16*)alloc((size_t)MROWS * NEMBD * 2);       // 16 MB

  // 1. convert inputs to f16
  {
    const int nx = MROWS * NEMBD;             // 8388608
    const int nw = NEMBD * 3 * NEMBD;         // 3145728
    const int np = NEMBD * NEMBD;             // 1048576
    cvt_f32_f16_kernel<<<nx / 1024, 256, 0, stream>>>(x, xh, nx);
    cvt_f32_f16_kernel<<<nw / 1024, 256, 0, stream>>>(Wqkv, wqkvh, nw);
    cvt_f32_f16_kernel<<<np / 1024, 256, 0, stream>>>(Wproj, wprjh, np);
  }

  // 2. QKV projection: [8192,1024] x [1024,3072]
  gemm_f16_kernel<false><<<dim3(MROWS / 128, (3 * NEMBD) / 64), 256, 0, stream>>>(
      xh, wqkvh, bqkv, qkvh, MROWS, 3 * NEMBD, NEMBD);

  // 3. RoPE + head split
  rope_pack_kernel<<<(MROWS * NEMBD / 2) / 256, 256, 0, stream>>>(qkvh, qh, kh, vh);

  // 4. Flash attention
  attn_kernel<<<dim3(4 * NHEAD, NCTX / 128), 256, 0, stream>>>(qh, kh, vh, amask, oh);

  // 5. Output projection -> f32 d_out
  gemm_f16_kernel<true><<<dim3(MROWS / 128, NEMBD / 64), 256, 0, stream>>>(
      oh, wprjh, bproj, d_out, MROWS, NEMBD, NEMBD);
}
